// SSM_35751307772688
// MI455X (gfx1250) — matrix operand order repaired and checked by reading, not compiled, not run on hardware
//
#include <hip/hip_runtime.h>
#include <hip/hip_bf16.h>
#include <math.h>

// ---------------- problem constants ----------------
#define NHEADS   64
#define EMB      2048
#define STATE    128
#define KCONV    4
#define HEAD_DIM 64
#define CHUNK    128
#define INTER    4096
#define CONV_DIM 4352
#define PROJ     8512
#define BATCH    2
#define SEQ      2048
#define NTOK     (BATCH*SEQ)      // 4096
#define NCH      (SEQ/CHUNK)      // 16

typedef __bf16 bf16_t;
typedef __attribute__((ext_vector_type(16))) __bf16 v16bf;
typedef __attribute__((ext_vector_type(8)))  float  v8f;
typedef __attribute__((ext_vector_type(4)))  int    i32x4_t;

#define AS1 __attribute__((address_space(1)))
#define AS3 __attribute__((address_space(3)))

#if defined(__HIP_DEVICE_COMPILE__) && __has_builtin(__builtin_amdgcn_global_load_async_to_lds_b128)
#define HAS_ASYNC_LDS 1
#else
#define HAS_ASYNC_LDS 0
#endif

__device__ __forceinline__ void wait_async0() {
#if __has_builtin(__builtin_amdgcn_s_wait_asynccnt)
  __builtin_amdgcn_s_wait_asynccnt(0);
#else
  asm volatile("s_wait_asynccnt 0x0" ::: "memory");
#endif
}

// POD 16-byte chunk for vector loads (HIP uint4 has non-trivial ctors -> no union)
struct __align__(16) Q16 { unsigned int a, b, c, d; };
union FragU { v16bf v; Q16 q[2]; };

// A fragment (16xK tile, 16-bit): lane m = lane&15, K = 8*(lane>>4) + {0..7, 16..23}
__device__ __forceinline__ v16bf frag_a_load(const bf16_t* p, int ld, int k0) {
  int lane = threadIdx.x & 31;
  int m = lane & 15, g = lane >> 4;
  const bf16_t* r = p + (size_t)m * ld + k0 + 8 * g;
  FragU f;
  f.q[0] = *reinterpret_cast<const Q16*>(r);
  f.q[1] = *reinterpret_cast<const Q16*>(r + 16);
  return f.v;
}

// B fragment (Kx16 tile, 16-bit), stored as 16 "column rows" of length ld with K contiguous:
// lane n = lane&15, K = 8*(lane>>4) + {0..7, 16..23}
__device__ __forceinline__ v16bf frag_b_load(const bf16_t* p, int ld, int k0) {
  int lane = threadIdx.x & 31;
  int n = lane & 15, g = lane >> 4;
  const bf16_t* r = p + (size_t)n * ld + k0 + 8 * g;
  FragU f;
  f.q[0] = *reinterpret_cast<const Q16*>(r);
  f.q[1] = *reinterpret_cast<const Q16*>(r + 16);
  return f.v;
}

#define WMMA_BF16(a, b, c) \
  __builtin_amdgcn_wmma_f32_16x16x32_bf16(false, (a), false, (b), (short)0, (c), false, false)

__device__ __forceinline__ float silu_f(float x) { return x / (1.f + expf(-x)); }

// ---------------- elementwise helpers ----------------
__global__ void __launch_bounds__(256)
k_f32_to_bf16(const float* __restrict__ in, bf16_t* __restrict__ out, long n) {
  long i = (long)blockIdx.x * blockDim.x + threadIdx.x;
  if (i < n) out[i] = (bf16_t)in[i];
}

// ---------------- GEMM: C[M,N] = A[M,K](bf16) * W[N,K](bf16)^T, fp32 out ----------------
// block = 256 thr = 8 waves; block tile 256Mx64N; wave tile 32Mx64N; K-step 32.
// B (weight) tile is shared by all 8 waves -> stage it through LDS with CDNA5
// async global->LDS copies, double buffered so the DMA of step k+1 overlaps the
// WMMAs of step k. Falls back to direct global B loads if the builtin is absent.
__global__ void __launch_bounds__(256)
k_gemm_bf16(const bf16_t* __restrict__ A, const bf16_t* __restrict__ W,
            float* __restrict__ C, int M, int N, int K) {
  int w = threadIdx.x >> 5;
  int m0 = blockIdx.y * 256 + w * 32;
  int n0 = blockIdx.x * 64;
  const bf16_t* a0 = A + (size_t)m0 * K;
  const bf16_t* a1 = A + (size_t)(m0 + 16) * K;
  v8f acc0[4] = {}, acc1[4] = {};
#if HAS_ASYNC_LDS
  __shared__ bf16_t Bs[2][64][32];   // 8 KB, double buffered
  int tid = threadIdx.x;
  int brow = tid >> 2;               // 0..63 : weight row (output col)
  int bseg = (tid & 3) * 8;          // 16-byte segment within 32-elem K slice
  bf16_t* gsrc = const_cast<bf16_t*>(W) + (size_t)(n0 + brow) * K + bseg;
  __builtin_amdgcn_global_load_async_to_lds_b128(
      (AS1 i32x4_t*)gsrc, (AS3 i32x4_t*)&Bs[0][brow][bseg], 0, 0);
  wait_async0();
  __syncthreads();
  for (int k0 = 0; k0 < K; k0 += 32) {
    int buf = (k0 >> 5) & 1;
    if (k0 + 32 < K) {  // block-uniform: no divergence at WMMA
      __builtin_amdgcn_global_load_async_to_lds_b128(
          (AS1 i32x4_t*)(gsrc + k0 + 32), (AS3 i32x4_t*)&Bs[buf ^ 1][brow][bseg], 0, 0);
    }
    v16bf fa0 = frag_a_load(a0, K, k0);
    v16bf fa1 = frag_a_load(a1, K, k0);
#pragma unroll
    for (int nt = 0; nt < 4; ++nt) {
      v16bf fb = frag_b_load(&Bs[buf][nt * 16][0], 32, 0);
      acc0[nt] = WMMA_BF16(fa0, fb, acc0[nt]);
      acc1[nt] = WMMA_BF16(fa1, fb, acc1[nt]);
    }
    wait_async0();
    __syncthreads();
  }
#else
  for (int k0 = 0; k0 < K; k0 += 32) {
    v16bf fa0 = frag_a_load(a0, K, k0);
    v16bf fa1 = frag_a_load(a1, K, k0);
#pragma unroll
    for (int nt = 0; nt < 4; ++nt) {
      v16bf fb = frag_b_load(W + (size_t)(n0 + nt * 16) * K, K, k0);
      acc0[nt] = WMMA_BF16(fa0, fb, acc0[nt]);
      acc1[nt] = WMMA_BF16(fa1, fb, acc1[nt]);
    }
  }
#endif
  int lane = threadIdx.x & 31;
  int mo = 8 * (lane >> 4), nn = lane & 15;
#pragma unroll
  for (int nt = 0; nt < 4; ++nt) {
#pragma unroll
    for (int r = 0; r < 8; ++r) {
      C[(size_t)(m0 + r + mo) * N + n0 + nt * 16 + nn]      = acc0[nt][r];
      C[(size_t)(m0 + 16 + r + mo) * N + n0 + nt * 16 + nn] = acc1[nt][r];
    }
  }
}

// ---------------- depthwise causal conv + SiLU ----------------
__global__ void __launch_bounds__(256)
k_conv_silu(const float* __restrict__ proj, const float* __restrict__ cw,
            const float* __restrict__ cb, float* __restrict__ hbc) {
  long i = (long)blockIdx.x * blockDim.x + threadIdx.x;
  if (i >= (long)NTOK * CONV_DIM) return;
  int ch = (int)(i % CONV_DIM);
  long tok = i / CONV_DIM;
  int t = (int)(tok % SEQ);
  float acc = cb[ch];
#pragma unroll
  for (int k = 0; k < KCONV; ++k) {
    int tt = t - (KCONV - 1) + k;
    if (tt >= 0) acc += proj[(tok + (tt - t)) * PROJ + INTER + ch] * cw[ch * KCONV + k];
  }
  hbc[i] = silu_f(acc);
}

// ---------------- dt softplus + A*dt ----------------
__global__ void __launch_bounds__(256)
k_dt(const float* __restrict__ proj, const float* __restrict__ dt_bias,
     const float* __restrict__ A_log, float* __restrict__ dtsp, float* __restrict__ aS) {
  long i = (long)blockIdx.x * blockDim.x + threadIdx.x;
  if (i >= (long)NTOK * NHEADS) return;
  int h = (int)(i & (NHEADS - 1));
  long tok = i >> 6;
  int b = (int)(tok / SEQ), s = (int)(tok % SEQ);
  float v = proj[tok * PROJ + INTER + CONV_DIM + h] + dt_bias[h];
  float sp = (v > 20.f) ? v : log1pf(expf(v));
  dtsp[i] = sp;
  aS[((size_t)(b * NHEADS + h)) * SEQ + s] = -expf(A_log[h]) * sp;
}

// ---------------- per-chunk inclusive cumsum (128 wide) ----------------
__global__ void __launch_bounds__(128)
k_cumsum(const float* __restrict__ aS, float* __restrict__ acum) {
  int c = blockIdx.x % NCH;
  int bh = blockIdx.x / NCH;
  __shared__ float s[CHUNK];
  int l = threadIdx.x;
  size_t base = (size_t)bh * SEQ + (size_t)c * CHUNK;
  s[l] = aS[base + l];
  __syncthreads();
  for (int o = 1; o < CHUNK; o <<= 1) {
    float v = (l >= o) ? s[l - o] : 0.f;
    __syncthreads();
    s[l] += v;
    __syncthreads();
  }
  acum[base + l] = s[l];
}

// ---------------- prep: hid*dt -> bf16; split B,C -> bf16 ----------------
__global__ void __launch_bounds__(256)
k_prep(const float* __restrict__ hbc, const float* __restrict__ dtsp,
       bf16_t* __restrict__ hidS, bf16_t* __restrict__ Bb, bf16_t* __restrict__ Cb) {
  long i = (long)blockIdx.x * blockDim.x + threadIdx.x;
  if (i >= (long)NTOK * CONV_DIM) return;
  int ch = (int)(i % CONV_DIM);
  long tok = i / CONV_DIM;
  float v = hbc[i];
  if (ch < INTER) {
    int h = ch >> 6;
    hidS[tok * INTER + ch] = (bf16_t)(v * dtsp[tok * NHEADS + h]);
  } else if (ch < INTER + STATE) {
    Bb[tok * STATE + (ch - INTER)] = (bf16_t)v;
  } else {
    Cb[tok * STATE + (ch - INTER - STATE)] = (bf16_t)v;
  }
}

// ---------------- SSM chunk-local part 1: G, Y_diag, states ----------------
__global__ void __launch_bounds__(256)
k_ssm1(const bf16_t* __restrict__ hidS, const bf16_t* __restrict__ Bb,
       const bf16_t* __restrict__ Cbuf, const float* __restrict__ acum,
       float* __restrict__ y, float* __restrict__ states_c) {
  int h = blockIdx.x % NHEADS;
  int c = (blockIdx.x / NHEADS) % NCH;
  int b = blockIdx.x / (NHEADS * NCH);
  int tid = threadIdx.x;
  int w = tid >> 5;
  int lane = tid & 31;
  int g = lane >> 4, nn = lane & 15;

  __shared__ float a_cum_s[CHUNK];
  __shared__ float decay_s[CHUNK];
  __shared__ bf16_t hidT_p[HEAD_DIM][136];   // hid_c^T         (B-operand for Y_diag)
  __shared__ bf16_t hidT_d[HEAD_DIM][136];   // (hid_c*decay)^T (A-operand for states)
  __shared__ bf16_t BT[STATE][136];          // B^T             (B-operand for states)
  __shared__ bf16_t GL[8][16][136];          // G*L rows per wave (A-operand for Y_diag)

  size_t tok0 = (size_t)b * SEQ + (size_t)c * CHUNK;
  if (tid < CHUNK) a_cum_s[tid] = acum[((size_t)(b * NHEADS + h)) * SEQ + c * CHUNK + tid];
  __syncthreads();
  if (tid < CHUNK) decay_s[tid] = expf(a_cum_s[CHUNK - 1] - a_cum_s[tid]);
  __syncthreads();

  for (int idx = tid; idx < CHUNK * HEAD_DIM; idx += 256) {
    int l = idx >> 6, p = idx & 63;
    float v = (float)hidS[(tok0 + l) * INTER + (size_t)h * HEAD_DIM + p];
    hidT_p[p][l] = (bf16_t)v;
    hidT_d[p][l] = (bf16_t)(v * decay_s[l]);
  }
  for (int idx = tid; idx < CHUNK * STATE; idx += 256) {
    int l = idx >> 7, n = idx & 127;
    BT[n][l] = Bb[(tok0 + l) * STATE + n];
  }
  __syncthreads();

  // ---- G = C * B^T  (wave w owns rows lbase..lbase+15), then scale by L, park in LDS
  int lbase = w * 16;
  const bf16_t* Crow = Cbuf + (tok0 + lbase) * STATE;
  const bf16_t* Brow = Bb + tok0 * STATE;
  v8f gacc[8] = {};
  for (int kk = 0; kk < 4; ++kk) {
    v16bf fa = frag_a_load(Crow, STATE, kk * 32);
#pragma unroll
    for (int nt = 0; nt < 8; ++nt) {
      v16bf fb = frag_b_load(Brow + (size_t)(nt * 16) * STATE, STATE, kk * 32);
      gacc[nt] = WMMA_BF16(fa, fb, gacc[nt]);
    }
  }
#pragma unroll
  for (int nt = 0; nt < 8; ++nt) {
#pragma unroll
    for (int r = 0; r < 8; ++r) {
      int ml = r + 8 * g;
      int l = lbase + ml;
      int s = nt * 16 + nn;
      float v = (s <= l) ? gacc[nt][r] * expf(a_cum_s[l] - a_cum_s[s]) : 0.f;
      GL[w][ml][s] = (bf16_t)v;
    }
  }

  // ---- Y_diag = (G*L) * hid_c   (A from LDS GL, B from LDS hidT_p)
  v8f yacc[4] = {};
  for (int kk = 0; kk < 4; ++kk) {
    v16bf fa = frag_a_load(&GL[w][0][0], 136, kk * 32);
#pragma unroll
    for (int pt = 0; pt < 4; ++pt) {
      v16bf fb = frag_b_load(&hidT_p[pt * 16][0], 136, kk * 32);
      yacc[pt] = WMMA_BF16(fa, fb, yacc[pt]);
    }
  }
#pragma unroll
  for (int pt = 0; pt < 4; ++pt)
#pragma unroll
    for (int r = 0; r < 8; ++r) {
      int l = lbase + r + 8 * g;
      y[(tok0 + l) * INTER + (size_t)h * HEAD_DIM + pt * 16 + nn] = yacc[pt][r];
    }

  // ---- states[p,n] = sum_l (hid_c*decay)[l,p] * B[l,n]
  int pstrip = (w & 3) * 16;
  int nb = (w >> 2) * 64;
  v8f sacc[4] = {};
  for (int kk = 0; kk < 4; ++kk) {
    v16bf fa = frag_a_load(&hidT_d[pstrip][0], 136, kk * 32);
#pragma unroll
    for (int nt = 0; nt < 4; ++nt) {
      v16bf fb = frag_b_load(&BT[nb + nt * 16][0], 136, kk * 32);
      sacc[nt] = WMMA_BF16(fa, fb, sacc[nt]);
    }
  }
  size_t sb = ((size_t)(b * NCH + c) * NHEADS + h) * HEAD_DIM;
#pragma unroll
  for (int nt = 0; nt < 4; ++nt)
#pragma unroll
    for (int r = 0; r < 8; ++r) {
      int p = pstrip + r + 8 * g;
      states_c[(sb + p) * STATE + nb + nt * 16 + nn] = sacc[nt][r];
    }
}

// ---------------- cross-chunk state recurrence (sequential over 16 chunks) ----------------
__global__ void __launch_bounds__(256)
k_rec(const float* __restrict__ states_c, const float* __restrict__ acum,
      bf16_t* __restrict__ states_prev) {
  int h = blockIdx.x % NHEADS;
  int b = blockIdx.x / NHEADS;
  int tid = threadIdx.x;
  float S[32];
#pragma unroll
  for (int i = 0; i < 32; ++i) S[i] = 0.f;
  for (int c = 0; c < NCH; ++c) {
    size_t base = ((size_t)(b * NCH + c) * NHEADS + h) * (HEAD_DIM * STATE);
#pragma unroll
    for (int i = 0; i < 32; ++i) states_prev[base + tid + i * 256] = (bf16_t)S[i];
    float sc = expf(acum[((size_t)(b * NHEADS + h)) * SEQ + c * CHUNK + (CHUNK - 1)]);
#pragma unroll
    for (int i = 0; i < 32; ++i) S[i] = S[i] * sc + states_c[base + tid + i * 256];
  }
}

// ---------------- SSM part 2: Y_off + D residual ----------------
__global__ void __launch_bounds__(256)
k_ssm2(const bf16_t* __restrict__ Cbuf, const bf16_t* __restrict__ states_prev,
       const float* __restrict__ acum, const float* __restrict__ hbc,
       const float* __restrict__ Dv, float* __restrict__ y) {
  int h = blockIdx.x % NHEADS;
  int c = (blockIdx.x / NHEADS) % NCH;
  int b = blockIdx.x / (NHEADS * NCH);
  int tid = threadIdx.x;
  int w = tid >> 5;
  int lane = tid & 31;
  int g = lane >> 4, nn = lane & 15;

  __shared__ float a_cum_s[CHUNK];
  size_t tok0 = (size_t)b * SEQ + (size_t)c * CHUNK;
  if (tid < CHUNK) a_cum_s[tid] = acum[((size_t)(b * NHEADS + h)) * SEQ + c * CHUNK + tid];
  __syncthreads();

  int lbase = w * 16;
  const bf16_t* Crow = Cbuf + (tok0 + lbase) * STATE;
  const bf16_t* Sb = states_prev + ((size_t)(b * NCH + c) * NHEADS + h) * (HEAD_DIM * STATE);
  v8f acc[4] = {};
  for (int kk = 0; kk < 4; ++kk) {
    v16bf fa = frag_a_load(Crow, STATE, kk * 32);
#pragma unroll
    for (int pt = 0; pt < 4; ++pt) {
      v16bf fb = frag_b_load(Sb + (size_t)(pt * 16) * STATE, STATE, kk * 32);
      acc[pt] = WMMA_BF16(fa, fb, acc[pt]);
    }
  }
  float Dh = Dv[h];
#pragma unroll
  for (int pt = 0; pt < 4; ++pt)
#pragma unroll
    for (int r = 0; r < 8; ++r) {
      int l = lbase + r + 8 * g;
      int p = pt * 16 + nn;
      size_t tok = tok0 + l;
      size_t yi = tok * INTER + (size_t)h * HEAD_DIM + p;
      float dres = Dh * hbc[tok * CONV_DIM + (size_t)h * HEAD_DIM + p];
      y[yi] = y[yi] + acc[pt][r] * expf(a_cum_s[l]) + dres;
    }
}

// ---------------- gated RMSNorm -> bf16 ----------------
__global__ void __launch_bounds__(256)
k_gate_norm(const float* __restrict__ y, const float* __restrict__ proj,
            const float* __restrict__ norm_w, bf16_t* __restrict__ outb) {
  long tok = blockIdx.x;
  __shared__ float red[256];
  const float* yp = y + (size_t)tok * INTER;
  const float* gp = proj + (size_t)tok * PROJ;
  float ss = 0.f;
  for (int i = threadIdx.x; i < INTER; i += 256) {
    float v = yp[i] * silu_f(gp[i]);
    ss += v * v;
  }
  red[threadIdx.x] = ss;
  __syncthreads();
  for (int o = 128; o > 0; o >>= 1) {
    if (threadIdx.x < o) red[threadIdx.x] += red[threadIdx.x + o];
    __syncthreads();
  }
  float rs = rsqrtf(red[0] / (float)INTER + 1e-6f);
  for (int i = threadIdx.x; i < INTER; i += 256) {
    float v = yp[i] * silu_f(gp[i]);
    outb[(size_t)tok * INTER + i] = (bf16_t)(norm_w[i] * v * rs);
  }
}

// ---------------- workspace layout (bytes) ----------------
static constexpr size_t SZ_XB   = (size_t)NTOK * EMB * 2;
static constexpr size_t SZ_WIN  = (size_t)PROJ * EMB * 2;
static constexpr size_t SZ_WOUT = (size_t)EMB * INTER * 2;
static constexpr size_t SZ_PROJ = (size_t)NTOK * PROJ * 4;
static constexpr size_t SZ_HBC  = (size_t)NTOK * CONV_DIM * 4;
static constexpr size_t SZ_DTSP = (size_t)NTOK * NHEADS * 4;
static constexpr size_t SZ_AS   = (size_t)BATCH * NHEADS * SEQ * 4;
static constexpr size_t SZ_ACUM = SZ_AS;
static constexpr size_t SZ_HIDS = (size_t)NTOK * INTER * 2;
static constexpr size_t SZ_BB   = (size_t)NTOK * STATE * 2;
static constexpr size_t SZ_CB   = SZ_BB;
static constexpr size_t SZ_Y    = (size_t)NTOK * INTER * 4;
static constexpr size_t SZ_STC  = (size_t)BATCH * NCH * NHEADS * HEAD_DIM * STATE * 4;
static constexpr size_t SZ_STP  = (size_t)BATCH * NCH * NHEADS * HEAD_DIM * STATE * 2;

static constexpr size_t OFF_XB   = 0;
static constexpr size_t OFF_WIN  = OFF_XB + SZ_XB;
static constexpr size_t OFF_WOUT = OFF_WIN + SZ_WIN;
static constexpr size_t OFF_PROJ = OFF_WOUT + SZ_WOUT;
static constexpr size_t OFF_HBC  = OFF_PROJ + SZ_PROJ;
static constexpr size_t OFF_DTSP = OFF_HBC + SZ_HBC;
static constexpr size_t OFF_AS   = OFF_DTSP + SZ_DTSP;
static constexpr size_t OFF_ACUM = OFF_AS + SZ_AS;
static constexpr size_t OFF_HIDS = OFF_ACUM + SZ_ACUM;
static constexpr size_t OFF_BB   = OFF_HIDS + SZ_HIDS;
static constexpr size_t OFF_CB   = OFF_BB + SZ_BB;
static constexpr size_t OFF_Y    = OFF_CB + SZ_CB;
static constexpr size_t OFF_STC  = OFF_Y + SZ_Y;
static constexpr size_t OFF_STP  = OFF_STC + SZ_STC;
static constexpr size_t OFF_GB   = OFF_STP + SZ_STP;

extern "C" void kernel_launch(void* const* d_in, const int* in_sizes, int n_in,
                              void* d_out, int out_size, void* d_ws, size_t ws_size,
                              hipStream_t stream) {
  const float* x       = (const float*)d_in[0];
  const float* w_in    = (const float*)d_in[1];
  const float* conv_w  = (const float*)d_in[2];
  const float* conv_b  = (const float*)d_in[3];
  const float* dt_bias = (const float*)d_in[4];
  const float* A_log   = (const float*)d_in[5];
  const float* Dv      = (const float*)d_in[6];
  const float* norm_w  = (const float*)d_in[7];
  const float* w_out   = (const float*)d_in[8];
  float* out = (float*)d_out;

  char* ws = (char*)d_ws;
  bf16_t* xb    = (bf16_t*)(ws + OFF_XB);
  bf16_t* winb  = (bf16_t*)(ws + OFF_WIN);
  bf16_t* woutb = (bf16_t*)(ws + OFF_WOUT);
  float*  proj  = (float*)(ws + OFF_PROJ);
  float*  hbc   = (float*)(ws + OFF_HBC);
  float*  dtsp  = (float*)(ws + OFF_DTSP);
  float*  aS    = (float*)(ws + OFF_AS);
  float*  acum  = (float*)(ws + OFF_ACUM);
  bf16_t* hidS  = (bf16_t*)(ws + OFF_HIDS);
  bf16_t* Bb    = (bf16_t*)(ws + OFF_BB);
  bf16_t* Cb    = (bf16_t*)(ws + OFF_CB);
  float*  y     = (float*)(ws + OFF_Y);
  float*  stc   = (float*)(ws + OFF_STC);
  bf16_t* stp   = (bf16_t*)(ws + OFF_STP);
  bf16_t* gb    = (bf16_t*)(ws + OFF_GB);

  const int T = 256;
  // convert fp32 inputs to bf16
  {
    long n = (long)NTOK * EMB;
    k_f32_to_bf16<<<(unsigned)((n + T - 1) / T), T, 0, stream>>>(x, xb, n);
    n = (long)PROJ * EMB;
    k_f32_to_bf16<<<(unsigned)((n + T - 1) / T), T, 0, stream>>>(w_in, winb, n);
    n = (long)EMB * INTER;
    k_f32_to_bf16<<<(unsigned)((n + T - 1) / T), T, 0, stream>>>(w_out, woutb, n);
  }
  // in_proj GEMM: proj[NTOK, PROJ]
  {
    dim3 grid(PROJ / 64, NTOK / 256);
    k_gemm_bf16<<<grid, T, 0, stream>>>(xb, winb, proj, NTOK, PROJ, EMB);
  }
  // conv + SiLU
  {
    long n = (long)NTOK * CONV_DIM;
    k_conv_silu<<<(unsigned)((n + T - 1) / T), T, 0, stream>>>(proj, conv_w, conv_b, hbc);
  }
  // dt / A*dt
  {
    long n = (long)NTOK * NHEADS;
    k_dt<<<(unsigned)((n + T - 1) / T), T, 0, stream>>>(proj, dt_bias, A_log, dtsp, aS);
  }
  // per-chunk cumsum
  k_cumsum<<<BATCH * NHEADS * NCH, 128, 0, stream>>>(aS, acum);
  // prep bf16 operands
  {
    long n = (long)NTOK * CONV_DIM;
    k_prep<<<(unsigned)((n + T - 1) / T), T, 0, stream>>>(hbc, dtsp, hidS, Bb, Cb);
  }
  // chunk-local: G, Y_diag, states
  k_ssm1<<<BATCH * NCH * NHEADS, T, 0, stream>>>(hidS, Bb, Cb, acum, y, stc);
  // cross-chunk recurrence
  k_rec<<<BATCH * NHEADS, T, 0, stream>>>(stc, acum, stp);
  // Y_off + D residual
  k_ssm2<<<BATCH * NCH * NHEADS, T, 0, stream>>>(Cb, stp, acum, hbc, Dv, y);
  // gated RMSNorm -> bf16
  k_gate_norm<<<NTOK, T, 0, stream>>>(y, proj, norm_w, gb);
  // out_proj GEMM -> d_out
  {
    dim3 grid(EMB / 64, NTOK / 256);
    k_gemm_bf16<<<grid, T, 0, stream>>>(gb, woutb, out, NTOK, EMB, INTER);
  }
}
